// HypConvHyperboloid_57732950393214
// MI455X (gfx1250) — matrix operand order repaired and checked by reading, not compiled, not run on hardware
//
#include <hip/hip_runtime.h>
#include <hip/hip_bf16.h>

// ---------------------------------------------------------------------------
// HypConv on the hyperboloid:  out = [sqrt(1+|z|^2), z],  z = W * hcat(x3x3)
// GEMM: M = 32*64*64 = 131072 pixels, K = 577 (pad 608), N = 255 (pad 256)
// bf16 WMMA with hi/lo split-precision (3 WMMAs per product) ~ fp32 accuracy.
// Blocking: 1 block = one full image row (64 pixels) x all 256 channels, so
// each B fragment pulled from L2 feeds 4 m-subtiles (64 FLOP / L2 byte).
// ---------------------------------------------------------------------------

typedef __attribute__((ext_vector_type(16))) __bf16 v16bf;
typedef __attribute__((ext_vector_type(8)))  float  v8f;

#define KTILES 19                        // ceil(577/32) -> K padded to 608
#define NTILES 16                        // 256 / 16
#define BPACK_ELEMS (KTILES * NTILES * 32 * 16)   // 155648 bf16 per (hi|lo)
#define MSUB 4                           // 4 m-subtiles of 16 -> 64 pixels/block

static __device__ __forceinline__ __bf16 f2bf(float f) {
  unsigned u = __builtin_bit_cast(unsigned, f);
  unsigned r = (u + 0x7FFFu + ((u >> 16) & 1u)) >> 16;   // round-nearest-even
  unsigned short s = (unsigned short)r;
  return __builtin_bit_cast(__bf16, s);
}
static __device__ __forceinline__ float bf2f(__bf16 h) {
  unsigned short s = __builtin_bit_cast(unsigned short, h);
  unsigned u = ((unsigned)s) << 16;
  return __builtin_bit_cast(float, u);
}

// ---------------------------------------------------------------------------
// Pack W (255 x 577, row-major) into WMMA B-matrix fragment order, bf16 hi/lo.
// Flat bf16 index: idx = (((kt*16 + nt)*32) + lane)*16 + e
//   B order here (k contiguous, as A): n = nt*16 + (lane&15)
//                                     k = kt*32 + 16*(lane>>4) + e
// ---------------------------------------------------------------------------
__global__ __launch_bounds__(256) void pack_W_kernel(const float* __restrict__ W,
                                                     __bf16* __restrict__ gBhi,
                                                     __bf16* __restrict__ gBlo) {
  int idx = blockIdx.x * 256 + threadIdx.x;
  if (idx >= BPACK_ELEMS) return;
  int e    = idx & 15;
  int lane = (idx >> 4) & 31;
  int nt   = (idx >> 9) & 15;
  int kt   = idx >> 13;
  int n = nt * 16 + (lane & 15);
  int k = kt * 32 + 16 * (lane >> 4) + e;
  float f = (n < 255 && k < 577) ? W[n * 577 + k] : 0.0f;
  __bf16 hi = f2bf(f);
  gBhi[idx] = hi;
  gBlo[idx] = f2bf(f - bf2f(hi));
}

// ---------------------------------------------------------------------------
// Main kernel: 1 block = 64 pixels (one image row) x 256 out channels.
// LDS layout (static, 152.3 KB; reused across phases with barriers):
//   [0      , 77824 )  A panel hi : (kt,ms,lane,e) bf16, 19*4*512 elems
//   [77824  , 155648)  A panel lo
//   [155648 , 155904)  t_cat (64 floats)
//   after GEMM (overlaps A panel, barrier-protected):
//   [0      , 65536 )  z buffer 64x256 f32
//   [65536  , 66560 )  256 partial sums
//   [66560  , 66816 )  64 t_out
// ---------------------------------------------------------------------------
#define AHI_OFF  0
#define ALO_OFF  77824
#define TC_OFF   155648
#define Z_OFF    0
#define PART_OFF 65536
#define TOUT_OFF 66560
#define SMEM_BYTES 155904
#define APANEL_ELEMS (KTILES * MSUB * 512)     // 38912 bf16 per (hi|lo)

__global__ __launch_bounds__(256) void hyp_conv_kernel(
    const float*  __restrict__ x,      // (32,64,64,65)
    const __bf16* __restrict__ gBhi,   // packed Wt hi
    const __bf16* __restrict__ gBlo,   // packed Wt lo
    const float*  __restrict__ bias,   // (255,)
    float*        __restrict__ out)    // (32,64,64,256)
{
  __shared__ __align__(16) unsigned char smem[SMEM_BYTES];
  __bf16* ldsAh = (__bf16*)(smem + AHI_OFF);
  __bf16* ldsAl = (__bf16*)(smem + ALO_OFF);
  float*  ldsTC = (float*)(smem + TC_OFF);

  const int t   = threadIdx.x;
  const int blk = blockIdx.x;          // 32*64 = 2048 blocks: one image row
  const int hh  = blk & 63;
  const int bb  = blk >> 6;
  const size_t rowbase = ((size_t)bb * 64 + hh) * 64;   // pixel index of w=0

  // ---- Phase 0: t_cat[w] = sqrt( sum_9 t_i^2 - 8 ), edge-clamped (x is L2 hot)
  if (t < 64) {
    float s = 0.0f;
    #pragma unroll
    for (int di = 0; di < 3; ++di) {
      int gh = min(max(hh + di - 1, 0), 63);
      #pragma unroll
      for (int dj = 0; dj < 3; ++dj) {
        int gw = min(max(t + dj - 1, 0), 63);
        float tv = x[(((size_t)bb * 64 + gh) * 64 + gw) * 65];
        s += tv * tv;
      }
    }
    ldsTC[t] = sqrtf(s - 8.0f);
  }
  __syncthreads();

  // ---- Phase 1: build 64x608 A panel in WMMA fragment order (hi/lo bf16)
  //   flat = ((kt*4 + ms)*32 + lane)*16 + e
  //   A order here (k contiguous, as B): m = ms*16 + (lane&15)  (= pixel w)
  //   k = kt*32 + 16*(lane>>4) + e
  for (int f = t; f < APANEL_ELEMS; f += 256) {
    int e  = f & 15;
    int ln = (f >> 4) & 31;
    int ms = (f >> 9) & 3;
    int kt = f >> 11;
    int lh = ln >> 4;
    int m  = ms * 16 + (ln & 15);
    int k  = kt * 32 + 16 * lh + e;
    float v = 0.0f;
    if (k < 577) {
      if (k == 0) {
        v = ldsTC[m];
      } else {
        int q  = k - 1;
        int nb = q >> 6;              // neighbor 0..8
        int c  = (q & 63) + 1;        // spatial channel 1..64
        int di = (nb * 86) >> 8;      // nb / 3
        int dj = nb - 3 * di;         // nb % 3
        int gh = min(max(hh + di - 1, 0), 63);
        int gw = min(max(m + dj - 1, 0), 63);
        v = x[(((size_t)bb * 64 + gh) * 64 + gw) * 65 + c];
      }
    }
    __bf16 hi = f2bf(v);
    ldsAh[f] = hi;
    ldsAl[f] = f2bf(v - bf2f(hi));
  }
  __syncthreads();

  // ---- Phase 2: GEMM main loop. Wave w owns N-tiles (2w, 2w+1), 4 m-subtiles.
  const int wave = t >> 5;
  const int lane = t & 31;
  const int lh   = lane >> 4;
  const int nt0  = wave * 2, nt1 = nt0 + 1;
  const int ncol = lane & 15;
  const int n0   = nt0 * 16 + ncol;
  const int n1   = nt1 * 16 + ncol;

  float bv0 = (n0 < 255) ? bias[n0] : 0.0f;
  float bv1 = (n1 < 255) ? bias[n1] : 0.0f;
  v8f acc[MSUB][2];
  #pragma unroll
  for (int ms = 0; ms < MSUB; ++ms)
    #pragma unroll
    for (int r = 0; r < 8; ++r) { acc[ms][0][r] = bv0; acc[ms][1][r] = bv1; }

  const v16bf* Ah = (const v16bf*)(smem + AHI_OFF);
  const v16bf* Al = (const v16bf*)(smem + ALO_OFF);
  const v16bf* Bh = (const v16bf*)gBhi;
  const v16bf* Bl = (const v16bf*)gBlo;

  for (int kt = 0; kt < KTILES; ++kt) {
    v16bf b0h = Bh[(kt * 16 + nt0) * 32 + lane];
    v16bf b0l = Bl[(kt * 16 + nt0) * 32 + lane];
    v16bf b1h = Bh[(kt * 16 + nt1) * 32 + lane];
    v16bf b1l = Bl[(kt * 16 + nt1) * 32 + lane];
    #pragma unroll
    for (int ms = 0; ms < MSUB; ++ms) {
      v16bf ah = Ah[(kt * MSUB + ms) * 32 + lane];
      v16bf al = Al[(kt * MSUB + ms) * 32 + lane];
      // split-precision fp32 emulation: hi*hi + hi*lo + lo*hi, fp32 accumulate
      acc[ms][0] = __builtin_amdgcn_wmma_f32_16x16x32_bf16(false, ah, false, b0h,
                                                 (short)0, acc[ms][0], false, false);
      acc[ms][0] = __builtin_amdgcn_wmma_f32_16x16x32_bf16(false, ah, false, b0l,
                                                 (short)0, acc[ms][0], false, false);
      acc[ms][0] = __builtin_amdgcn_wmma_f32_16x16x32_bf16(false, al, false, b0h,
                                                 (short)0, acc[ms][0], false, false);
      acc[ms][1] = __builtin_amdgcn_wmma_f32_16x16x32_bf16(false, ah, false, b1h,
                                                 (short)0, acc[ms][1], false, false);
      acc[ms][1] = __builtin_amdgcn_wmma_f32_16x16x32_bf16(false, ah, false, b1l,
                                                 (short)0, acc[ms][1], false, false);
      acc[ms][1] = __builtin_amdgcn_wmma_f32_16x16x32_bf16(false, al, false, b1h,
                                                 (short)0, acc[ms][1], false, false);
    }
  }

  // ---- Phase 3: dump z to LDS (reuses A-panel space; barrier first)
  __syncthreads();
  float* ldsZ = (float*)(smem + Z_OFF);
  #pragma unroll
  for (int ms = 0; ms < MSUB; ++ms)
    #pragma unroll
    for (int r = 0; r < 8; ++r) {
      int mr = ms * 16 + r + 8 * lh;     // C/D layout: VGPR r -> M = r + 8*half
      ldsZ[mr * 256 + nt0 * 16 + ncol] = acc[ms][0][r];
      ldsZ[mr * 256 + nt1 * 16 + ncol] = acc[ms][1][r];
    }
  __syncthreads();

  // ---- Phase 4/5: t_out = sqrt(1 + sum z^2) per pixel (two-level reduce)
  float* ldsP = (float*)(smem + PART_OFF);
  {
    int p = t >> 2, seg = t & 3;         // 64 pixels x 4 segments of 64
    float ps = 0.0f;
    #pragma unroll 16
    for (int j = 0; j < 64; ++j) {
      float v = ldsZ[p * 256 + seg * 64 + j];
      ps += v * v;
    }
    ldsP[t] = ps;
  }
  __syncthreads();
  float* ldsT = (float*)(smem + TOUT_OFF);
  if (t < 64) {
    float s = ldsP[t * 4] + ldsP[t * 4 + 1] + ldsP[t * 4 + 2] + ldsP[t * 4 + 3];
    ldsT[t] = sqrtf(1.0f + s);
  }
  __syncthreads();

  // ---- Phase 6: coalesced store: out[pix][0]=t_out, out[pix][1+o]=z[o]
  #pragma unroll 8
  for (int i = 0; i < 64; ++i) {
    int f  = i * 256 + t;
    int mr = f >> 8;
    int ch = f & 255;
    float v = (ch == 0) ? ldsT[mr] : ldsZ[mr * 256 + (ch - 1)];
    out[(rowbase + mr) * 256 + ch] = v;
  }
}

// ---------------------------------------------------------------------------
extern "C" void kernel_launch(void* const* d_in, const int* in_sizes, int n_in,
                              void* d_out, int out_size, void* d_ws, size_t ws_size,
                              hipStream_t stream) {
  (void)in_sizes; (void)n_in; (void)out_size; (void)ws_size;
  const float* x    = (const float*)d_in[0];   // (32,64,64,65) fp32
  const float* W    = (const float*)d_in[1];   // (255,577) fp32
  const float* bias = (const float*)d_in[2];   // (255,) fp32
  float*       out  = (float*)d_out;           // (32,64,64,256) fp32

  // workspace: packed W' (hi then lo), 2 * 155648 bf16 = 622,592 bytes
  __bf16* gBhi = (__bf16*)d_ws;
  __bf16* gBlo = gBhi + BPACK_ELEMS;

  pack_W_kernel<<<(BPACK_ELEMS + 255) / 256, 256, 0, stream>>>(W, gBhi, gBlo);
  hyp_conv_kernel<<<32 * 64, 256, 0, stream>>>(x, gBhi, gBlo, bias, out);
}
